// DecoderBlock_9088150798452
// MI455X (gfx1250) — matrix operand order repaired and checked by reading, not compiled, not run on hardware
//
#include <hip/hip_runtime.h>
#include <hip/hip_bf16.h>
#include <math.h>

#define Bb   16
#define NP   512
#define Nn   1024
#define Cc   1024
#define Hh   16
#define HD   64
#define HID  4096
#define MROWS (Bb*NP)   /* 8192 */

#define BK   64
#define LDT  72   /* padded LDS stride in halves: 144B rows, 16B aligned */

typedef __attribute__((ext_vector_type(16))) _Float16 v16h;
typedef __attribute__((ext_vector_type(8)))  float    v8f;
typedef __attribute__((ext_vector_type(4)))  int      v4i;

#ifndef __has_builtin
#define __has_builtin(x) 0
#endif
#if __has_builtin(__builtin_amdgcn_global_load_async_to_lds_b128)
#define HAVE_ASYNC_LDS 1
#else
#define HAVE_ASYNC_LDS 0
#endif

union FragH { v16h v; _Float16 h[16]; uint4 q[2]; };

// A-matrix 16x32 f16 fragment: halves 0..15 -> K = hi*16 + j (contiguous,
// matching load_fragB). 'base' points at K=0 of this row.
__device__ __forceinline__ v16h load_fragA(const _Float16* base, int hi) {
  FragH u;
  u.q[0] = *(const uint4*)(base + hi * 16);
  u.q[1] = *(const uint4*)(base + hi * 16 + 8);
  return u.v;
}
// B-matrix 32x16 f16 fragment: lanes 0-15 hold K=0..15, lanes 16-31 K=16..31.
__device__ __forceinline__ v16h load_fragB(const _Float16* base, int hi) {
  FragH u;
  u.q[0] = *(const uint4*)(base + hi * 16);
  u.q[1] = *(const uint4*)(base + hi * 16 + 8);
  return u.v;
}
__device__ __forceinline__ v8f wmma16(v16h a, v16h b, v8f c) {
  return __builtin_amdgcn_wmma_f32_16x16x32_f16(false, a, false, b, (short)0, c,
                                                false, false);
}

// 16B global -> LDS copy: async LDS-DMA (ASYNCcnt) if available, else uint4.
__device__ __forceinline__ void cp16(const _Float16* g, _Float16* l) {
#if HAVE_ASYNC_LDS
  __builtin_amdgcn_global_load_async_to_lds_b128(
      (__attribute__((address_space(1))) v4i*)g,
      (__attribute__((address_space(3))) v4i*)l, 0, 0);
#else
  *(uint4*)l = *(const uint4*)g;
#endif
}
__device__ __forceinline__ void cp_wait() {
#if HAVE_ASYNC_LDS
#if __has_builtin(__builtin_amdgcn_s_wait_asynccnt)
  __builtin_amdgcn_s_wait_asynccnt(0);
#else
  asm volatile("s_wait_asynccnt 0x0" ::: "memory");
#endif
#endif
}

// ---------------------------------------------------------------- utilities
__global__ void cvt_f32_f16_kernel(const float* __restrict__ in,
                                   _Float16* __restrict__ out, size_t n) {
  size_t i = (size_t)blockIdx.x * blockDim.x + threadIdx.x;
  if (i < n) out[i] = (_Float16)in[i];
}

// f32 [K][Nc] -> f16 transposed [Nc][K] (32x32 LDS tiles, coalesced both ways)
__global__ __launch_bounds__(256) void cvt_t_kernel(const float* __restrict__ in,
                                                    _Float16* __restrict__ out,
                                                    int K, int Nc) {
  __shared__ float t[32][33];
  const int bx = blockIdx.x * 32;  // Nc offset
  const int by = blockIdx.y * 32;  // K offset
  const int tx = threadIdx.x & 31, ty = threadIdx.x >> 5;
  for (int r = ty; r < 32; r += 8)
    t[r][tx] = in[(size_t)(by + r) * Nc + bx + tx];
  __syncthreads();
  for (int r = ty; r < 32; r += 8)
    out[(size_t)(bx + r) * K + by + tx] = (_Float16)t[tx][r];
}

// idx[b][i] = position of i-th true bit of update_mask row b (row-uniform NP trues)
__global__ void idx_kernel(const unsigned char* __restrict__ mask,
                           int* __restrict__ idx) {
  int b = blockIdx.x;
  if (threadIdx.x == 0) {
    int cnt = 0;
    for (int n = 0; n < Nn && cnt < NP; ++n)
      if (mask[b * Nn + n]) idx[b * NP + cnt++] = n;
  }
}

// LayerNorm row kernel: f32 in -> f16 out (one 256-thread block per row of C=1024)
__global__ __launch_bounds__(256) void ln_kernel(const float* __restrict__ x,
                                                 const float* __restrict__ g,
                                                 const float* __restrict__ bta,
                                                 _Float16* __restrict__ out) {
  __shared__ float s1[256], s2[256];
  const int row = blockIdx.x, tid = threadIdx.x;
  const float* xr = x + (size_t)row * Cc;
  float a = 0.f, b2 = 0.f;
  for (int c = tid; c < Cc; c += 256) { float v = xr[c]; a += v; b2 += v * v; }
  s1[tid] = a; s2[tid] = b2; __syncthreads();
  for (int off = 128; off > 0; off >>= 1) {
    if (tid < off) { s1[tid] += s1[tid + off]; s2[tid] += s2[tid + off]; }
    __syncthreads();
  }
  float mu = s1[0] * (1.f / Cc);
  float var = s2[0] * (1.f / Cc) - mu * mu;
  float rs = rsqrtf(var + 1e-5f);
  for (int c = tid; c < Cc; c += 256)
    out[(size_t)row * Cc + c] = (_Float16)((xr[c] - mu) * rs * g[c] + bta[c]);
}

// ---------------------------------------------------------------- WMMA GEMM
enum { EPI_QKV = 0, EPI_PROJ = 1, EPI_FC1 = 2, EPI_FC2 = 3 };

struct GemmP {
  const _Float16* A;      // [M,K] f16 row-major
  const _Float16* Wt;     // [Nc,K] f16 (pre-transposed weights)
  const float* bias;      // [Nc]
  const float* resid;     // f32 residual [M,Cc] (PROJ/FC2)
  float* out32;           // f32 out (PROJ/FC2)
  _Float16* out16;        // f16 out (FC1)
  const int* idx;         // QKV scatter
  _Float16* q16;          // [B,H,NP,HD]
  _Float16* kf;           // [B,H,N,HD]
  _Float16* vf;           // [B,H,N,HD]
  int M, Nc, K;
};

template <int EPI>
__global__ __launch_bounds__(256) void gemm_kernel(GemmP p) {
  __shared__ _Float16 As[128 * LDT];   // [row][k]
  __shared__ _Float16 Bs[128 * LDT];   // [n][k]
  const int tid = threadIdx.x;
  const int wave = tid >> 5, lane = tid & 31;
  const int wm = wave >> 2, wn = wave & 3;       // 2x4 wave grid
  const int l16 = lane & 15, hi = lane >> 4;
  const int bm = blockIdx.y * 128, bn = blockIdx.x * 128;
  v8f acc[4][2] = {};

  for (int k0 = 0; k0 < p.K; k0 += BK) {
    const _Float16* Ag = p.A + (size_t)bm * p.K + k0;
    const _Float16* Wg = p.Wt + (size_t)bn * p.K + k0;
    // 128 rows x 64 halves each = 1024 vec8 copies per matrix
    for (int v = tid; v < 1024; v += 256) {
      int r = v >> 3, c = (v & 7) * 8;
      cp16(Ag + (size_t)r * p.K + c, As + r * LDT + c);
      cp16(Wg + (size_t)r * p.K + c, Bs + r * LDT + c);
    }
    cp_wait();
    __syncthreads();
#pragma unroll
    for (int c = 0; c < 2; ++c) {   // two K=32 chunks -> 16 WMMAs per stage
      v16h af[4], bf[2];
#pragma unroll
      for (int mi = 0; mi < 4; ++mi)
        af[mi] = load_fragA(As + (wm * 64 + mi * 16 + l16) * LDT + c * 32, hi);
#pragma unroll
      for (int ni = 0; ni < 2; ++ni)
        bf[ni] = load_fragB(Bs + (wn * 32 + ni * 16 + l16) * LDT + c * 32, hi);
#pragma unroll
      for (int mi = 0; mi < 4; ++mi)
#pragma unroll
        for (int ni = 0; ni < 2; ++ni)
          acc[mi][ni] = wmma16(af[mi], bf[ni], acc[mi][ni]);
    }
    __syncthreads();
  }

  // C/D layout: N = lane&15, M = vgpr + 8*(lane>=16)
#pragma unroll
  for (int mi = 0; mi < 4; ++mi)
#pragma unroll
    for (int ni = 0; ni < 2; ++ni)
#pragma unroll
      for (int r = 0; r < 8; ++r) {
        int row = bm + wm * 64 + mi * 16 + hi * 8 + r;
        int col = bn + wn * 32 + ni * 16 + l16;
        float val = acc[mi][ni][r] + p.bias[col];
        if (EPI == EPI_QKV) {
          int which = col >> 10, rem = col & 1023;
          int h = rem >> 6, d = rem & 63;
          int b = row >> 9, i = row & 511;
          if (which == 0) {
            p.q16[(((size_t)(b * Hh + h)) * NP + i) * HD + d] =
                (_Float16)(val * 0.125f);   // q * HD^-0.5 pre-applied
          } else {
            int slot = p.idx[b * NP + i];
            _Float16* dst = (which == 1) ? p.kf : p.vf;
            dst[(((size_t)(b * Hh + h)) * Nn + slot) * HD + d] = (_Float16)val;
          }
        } else if (EPI == EPI_FC1) {
          val = 0.5f * val * (1.0f + erff(val * 0.70710678118654752f)); // exact GELU
          p.out16[(size_t)row * HID + col] = (_Float16)val;
        } else {  // PROJ / FC2: bias + residual -> f32
          val += p.resid[(size_t)row * Cc + col];
          p.out32[(size_t)row * Cc + col] = val;
        }
      }
}

// ------------------------------------------------------------- attention
// One block per (b, h, 16-query tile); 4 waves. QK^T and PV through WMMA.
__global__ __launch_bounds__(128) void attn_kernel(const _Float16* __restrict__ q16,
                                                   const _Float16* __restrict__ kf,
                                                   const _Float16* __restrict__ vf,
                                                   _Float16* __restrict__ o16) {
  __shared__ float    S[16 * 1024];       // scores f32   (64 KB)
  __shared__ _Float16 P[16 * 1024];       // exp() as f16 (32 KB)
  __shared__ float    Opart[4 * 16 * 64]; // per-wave partial O (16 KB)
  __shared__ float    red[16 * 8];
  __shared__ float    rowmax[16], rowsum[16];

  const int tid = threadIdx.x;
  const int wave = tid >> 5, lane = tid & 31;
  const int l16 = lane & 15, hi = lane >> 4;
  const int qt = blockIdx.x, h = blockIdx.y, b = blockIdx.z;
  const size_t bh = (size_t)(b * Hh + h);

  // Q tile fragments (A-matrix, rows = queries, K-dim = head dim, 2 chunks)
  const _Float16* Qb = q16 + (bh * NP + (size_t)qt * 16) * HD;
  v16h qf[2];
#pragma unroll
  for (int c = 0; c < 2; ++c)
    qf[c] = load_fragA(Qb + l16 * HD + c * 32, hi);

  // S = (q*scale) K^T ; each wave handles 16 of 64 key tiles
  const _Float16* Kb = kf + bh * Nn * HD;
  for (int nt = wave; nt < 64; nt += 4) {
    v8f s = {};
#pragma unroll
    for (int c = 0; c < 2; ++c) {
      v16h bfrag = load_fragB(Kb + (size_t)(nt * 16 + l16) * HD + c * 32, hi);
      s = wmma16(qf[c], bfrag, s);
    }
#pragma unroll
    for (int r = 0; r < 8; ++r)
      S[(hi * 8 + r) * 1024 + nt * 16 + l16] = s[r];
  }
  __syncthreads();

  // softmax (deferred 1/sum): 8 threads per row
  const int rrow = tid >> 3, rsub = tid & 7;
  float mx = -3.0e38f;
  for (int c = rsub; c < 1024; c += 8) mx = fmaxf(mx, S[rrow * 1024 + c]);
  red[rrow * 8 + rsub] = mx; __syncthreads();
  if (rsub == 0) {
    float m = -3.0e38f;
    for (int j = 0; j < 8; ++j) m = fmaxf(m, red[rrow * 8 + j]);
    rowmax[rrow] = m;
  }
  __syncthreads();
  float m = rowmax[rrow], sum = 0.f;
  for (int c = rsub; c < 1024; c += 8) {
    float e = expf(S[rrow * 1024 + c] - m);
    P[rrow * 1024 + c] = (_Float16)e;
    sum += e;
  }
  red[rrow * 8 + rsub] = sum; __syncthreads();
  if (rsub == 0) {
    float s = 0.f;
    for (int j = 0; j < 8; ++j) s += red[rrow * 8 + j];
    rowsum[rrow] = s;
  }
  __syncthreads();

  // O = P V ; each wave accumulates 8 of 32 key-chunks, partials in LDS
  const _Float16* Vb = vf + bh * Nn * HD;
  v8f oacc[4] = {};
  for (int kc = wave * 8; kc < wave * 8 + 8; ++kc) {
    v16h pf = load_fragA(P + l16 * 1024 + kc * 32, hi);
#pragma unroll
    for (int dt = 0; dt < 4; ++dt) {
      FragH u;
#pragma unroll
      for (int j = 0; j < 16; ++j)   // B col n = dim, K = key (strided gather)
        u.h[j] = Vb[(size_t)(kc * 32 + hi * 16 + j) * HD + dt * 16 + l16];
      oacc[dt] = wmma16(pf, u.v, oacc[dt]);
    }
  }
#pragma unroll
  for (int dt = 0; dt < 4; ++dt)
#pragma unroll
    for (int r = 0; r < 8; ++r)
      Opart[(wave * 16 + hi * 8 + r) * 64 + dt * 16 + l16] = oacc[dt][r];
  __syncthreads();

  for (int e = tid; e < 16 * 64; e += 128) {
    int mr = e >> 6, d = e & 63;
    float s = Opart[mr * 64 + d] + Opart[(16 + mr) * 64 + d] +
              Opart[(32 + mr) * 64 + d] + Opart[(48 + mr) * 64 + d];
    s /= rowsum[mr];
    o16[((size_t)b * NP + qt * 16 + mr) * Cc + h * HD + d] = (_Float16)s;
  }
}

// ---------------------------------------------------------------- launch
extern "C" void kernel_launch(void* const* d_in, const int* in_sizes, int n_in,
                              void* d_out, int out_size, void* d_ws, size_t ws_size,
                              hipStream_t stream) {
  (void)in_sizes; (void)n_in; (void)out_size; (void)ws_size;
  const float* x       = (const float*)d_in[0];
  const float* cache_k = (const float*)d_in[1];
  const float* cache_v = (const float*)d_in[2];
  const unsigned char* mask = (const unsigned char*)d_in[3];
  const float* qkv_w = (const float*)d_in[4];
  const float* qkv_b = (const float*)d_in[5];
  const float* proj_w = (const float*)d_in[6];
  const float* proj_b = (const float*)d_in[7];
  const float* n1_g = (const float*)d_in[8];
  const float* n1_b = (const float*)d_in[9];
  const float* n2_g = (const float*)d_in[10];
  const float* n2_b = (const float*)d_in[11];
  const float* fc1_w = (const float*)d_in[12];
  const float* fc1_b = (const float*)d_in[13];
  const float* fc2_w = (const float*)d_in[14];
  const float* fc2_b = (const float*)d_in[15];
  float* out = (float*)d_out;

  char* ws = (char*)d_ws;
  size_t off = 0;
  auto carve = [&](size_t bytes) -> char* {
    char* p = ws + off;
    off += (bytes + 255) & ~(size_t)255;
    return p;
  };
  _Float16* wqkv16 = (_Float16*)carve((size_t)Cc * 3 * Cc * 2);   // [3C][C]  6 MB
  _Float16* wproj16 = (_Float16*)carve((size_t)Cc * Cc * 2);      // [C][C]   2 MB
  _Float16* wfc116 = (_Float16*)carve((size_t)Cc * HID * 2);      // [HID][C] 8 MB
  _Float16* wfc216 = (_Float16*)carve((size_t)HID * Cc * 2);      // [C][HID] 8 MB
  _Float16* h1_16 = (_Float16*)carve((size_t)MROWS * Cc * 2);     // 16 MB (reused as o16)
  _Float16* q16 = (_Float16*)carve((size_t)Bb * Hh * NP * HD * 2);// 16 MB (reused as h2_16)
  _Float16* kfull = (_Float16*)carve((size_t)Bb * Hh * Nn * HD * 2); // 32 MB
  _Float16* vfull = (_Float16*)carve((size_t)Bb * Hh * Nn * HD * 2); // 32 MB
  float* x2 = (float*)carve((size_t)MROWS * Cc * 4);              // 32 MB
  _Float16* a16 = (_Float16*)carve((size_t)MROWS * HID * 2);      // 64 MB
  int* idx = (int*)carve((size_t)Bb * NP * 4);
  _Float16* o16 = h1_16;    // h1 dead after QKV GEMM
  _Float16* h2_16 = q16;    // q dead after attention

  // 1) weights -> f16 transposed [Nc][K]; caches -> f16
  cvt_t_kernel<<<dim3(3 * Cc / 32, Cc / 32), 256, 0, stream>>>(qkv_w, wqkv16, Cc, 3 * Cc);
  cvt_t_kernel<<<dim3(Cc / 32, Cc / 32), 256, 0, stream>>>(proj_w, wproj16, Cc, Cc);
  cvt_t_kernel<<<dim3(HID / 32, Cc / 32), 256, 0, stream>>>(fc1_w, wfc116, Cc, HID);
  cvt_t_kernel<<<dim3(Cc / 32, HID / 32), 256, 0, stream>>>(fc2_w, wfc216, HID, Cc);
  {
    size_t n = (size_t)Bb * Hh * Nn * HD;
    cvt_f32_f16_kernel<<<(unsigned)((n + 255) / 256), 256, 0, stream>>>(cache_k, kfull, n);
    cvt_f32_f16_kernel<<<(unsigned)((n + 255) / 256), 256, 0, stream>>>(cache_v, vfull, n);
  }

  // 2) scatter indices
  idx_kernel<<<Bb, 32, 0, stream>>>(mask, idx);

  // 3) LN1
  ln_kernel<<<MROWS, 256, 0, stream>>>(x, n1_g, n1_b, h1_16);

  // 4) QKV GEMM (+ split/scale/scatter epilogue)
  {
    GemmP p{};
    p.A = h1_16; p.Wt = wqkv16; p.bias = qkv_b;
    p.idx = idx; p.q16 = q16; p.kf = kfull; p.vf = vfull;
    p.M = MROWS; p.Nc = 3 * Cc; p.K = Cc;
    gemm_kernel<EPI_QKV><<<dim3(3 * Cc / 128, MROWS / 128), 256, 0, stream>>>(p);
  }

  // 5) attention
  attn_kernel<<<dim3(NP / 16, Hh, Bb), 128, 0, stream>>>(q16, kfull, vfull, o16);

  // 6) proj GEMM + residual(x) -> x2 (f32)
  {
    GemmP p{};
    p.A = o16; p.Wt = wproj16; p.bias = proj_b; p.resid = x; p.out32 = x2;
    p.M = MROWS; p.Nc = Cc; p.K = Cc;
    gemm_kernel<EPI_PROJ><<<dim3(Cc / 128, MROWS / 128), 256, 0, stream>>>(p);
  }

  // 7) LN2
  ln_kernel<<<MROWS, 256, 0, stream>>>(x2, n2_g, n2_b, h2_16);

  // 8) FC1 GEMM + GELU -> a16
  {
    GemmP p{};
    p.A = h2_16; p.Wt = wfc116; p.bias = fc1_b; p.out16 = a16;
    p.M = MROWS; p.Nc = HID; p.K = Cc;
    gemm_kernel<EPI_FC1><<<dim3(HID / 128, MROWS / 128), 256, 0, stream>>>(p);
  }

  // 9) FC2 GEMM + residual(x2) -> out (d_out f32)
  {
    GemmP p{};
    p.A = a16; p.Wt = wfc216; p.bias = fc2_b; p.resid = x2; p.out32 = out;
    p.M = MROWS; p.Nc = Cc; p.K = HID;
    gemm_kernel<EPI_FC2><<<dim3(Cc / 128, MROWS / 128), 256, 0, stream>>>(p);
  }
}
